// Model_82317343195221
// MI455X (gfx1250) — compile-verified, matrix operand order checked by reading, not run on hardware
//
#include <hip/hip_runtime.h>
#include <hip/hip_bf16.h>

// ---------------- problem constants ----------------
#define BATCH   64
#define SEQL    1024
#define CH      512
#define HID     128
#define PRED    336
#define MAKER   25      // moving-average kernel
#define HALFW   12      // (MAKER-1)/2

typedef __attribute__((ext_vector_type(16))) __bf16 v16bf;
typedef __attribute__((ext_vector_type(8)))  float  v8f;

// ---------------- helpers ----------------
__device__ __forceinline__ unsigned short f2bf(float f) {
    unsigned int u = __builtin_bit_cast(unsigned int, f);
    u += 0x7FFFu + ((u >> 16) & 1u);            // round-to-nearest-even
    return (unsigned short)(u >> 16);
}

// CDNA5 async copy: 16 bytes per lane, global -> LDS, tracked by ASYNCcnt.
// VDST = VGPR with LDS byte address, VADDR = 64-bit global address (GV mode).
__device__ __forceinline__ void async_g2l_b128(unsigned lds_off, const void* gaddr) {
    asm volatile("global_load_async_to_lds_b128 %0, %1, off"
                 :: "v"(lds_off), "v"(gaddr)
                 : "memory");
}
__device__ __forceinline__ void wait_asynccnt0() {
    asm volatile("s_wait_asynccnt 0" ::: "memory");
}
__device__ __forceinline__ unsigned lds_addr32(const void* p) {
    // LDS aperture keeps the wave-relative LDS offset in the low 32 bits (ISA 10.2)
    return (unsigned)(size_t)p;
}

// Load one 16x(K=32) bf16 tile in the CDNA5 WMMA A/B register layout.
// Memory is row-major (row = M or N index, K contiguous), ld in elements.
// Per ISA 7.12.2 (16-bit A 16x32): lane L holds row (L&15); VGPR v holds the
// packed K-pair at k = 2*(v&3) + 16*(v>>2) + 8*(L>=16).
__device__ __forceinline__ v16bf load_tile(const unsigned short* __restrict__ base,
                                           int ld, int row0, int k0, int lane) {
    union { v16bf v; unsigned int u[8]; } t;
    const int r = row0 + (lane & 15);
    const int khalf = ((lane >> 4) & 1) * 8;
    const unsigned short* p = base + (size_t)r * (size_t)ld + (size_t)k0;
#pragma unroll
    for (int v = 0; v < 8; ++v) {
        const int k = ((v & 3) * 2) + ((v >> 2) * 16) + khalf;
        t.u[v] = *(const unsigned int*)(p + k);
    }
    return t.v;
}

// ---------------- kernel 0: f32 -> bf16 ----------------
__global__ void k_cvt_bf16(const float* __restrict__ src, unsigned short* __restrict__ dst, int n) {
    int i = blockIdx.x * blockDim.x + threadIdx.x;
    if (i < n) dst[i] = f2bf(src[i]);
}

// ---------------- kernel 1: instance-norm + series decomposition ----------------
// grid (CH/256, BATCH), block 256. Thread owns one (b,c): coalesced over c.
// seasonal stored (c*BATCH+b)-major (for xproj GEMM), trend stored (b*CH+c)-major.
__global__ void k_norm_decomp(const float* __restrict__ x,
                              float* __restrict__ means, float* __restrict__ stdev,
                              unsigned short* __restrict__ seas,
                              unsigned short* __restrict__ trend) {
    const int c = blockIdx.x * blockDim.x + threadIdx.x;
    const int b = blockIdx.y;
    const float* xp = x + (size_t)b * SEQL * CH + c;

    float s = 0.f, ss = 0.f;
    for (int l = 0; l < SEQL; ++l) {
        float v = xp[(size_t)l * CH];
        s += v; ss += v * v;
    }
    const float mean = s * (1.f / SEQL);
    const float var  = ss * (1.f / SEQL) - mean * mean;
    const float sd   = sqrtf(var + 1e-5f);
    means[b * CH + c] = mean;
    stdev[b * CH + c] = sd;
    const float inv = 1.f / sd;

    // replicate-padded sliding window sum over normalized series
    auto xn = [&](int j) { return (xp[(size_t)j * CH] - mean) * inv; };
    float w = (float)HALFW * xn(0);
    for (int j = 0; j <= HALFW; ++j) w += xn(j);

    unsigned short* sp = seas  + ((size_t)c * BATCH + b) * SEQL;
    unsigned short* tp = trend + ((size_t)b * CH + c) * SEQL;
    for (int l = 0; l < SEQL; ++l) {
        const float t = w * (1.f / (float)MAKER);
        const float v = xn(l);
        sp[l] = f2bf(v - t);
        tp[l] = f2bf(t);
        const int ia = (l + HALFW + 1 > SEQL - 1) ? SEQL - 1 : l + HALFW + 1;
        const int is = (l - HALFW < 0) ? 0 : l - HALFW;
        w += xn(ia) - xn(is);
    }
}

// ---------------- kernel 2: xproj GEMM (seasonal @ W_ih^T + b_ih) ----------------
// Block = one 16-row M-strip (contiguous 32 KB of A), async-staged to LDS once.
// 8 waves cover all 8 N-tiles (HID=128). M = CH*BATCH = 32768 -> 2048 blocks.
__global__ void __launch_bounds__(256) k_xproj(const unsigned short* __restrict__ A,   // [32768,1024]
                                               const unsigned short* __restrict__ Wih, // [128,1024]
                                               const float* __restrict__ b_ih,
                                               float* __restrict__ xproj) {            // [32768,128]
    __shared__ unsigned short As[16 * SEQL];     // 32 KB A strip
    const int lane = threadIdx.x & 31;
    const int wave = threadIdx.x >> 5;
    const int mt = blockIdx.x;                   // 0..2047

    // async stage: 32 KB = 8 waves x 8 issues x 32 lanes x 16 B, flat copy
    {
        const char* gsrc = (const char*)(A + (size_t)mt * 16 * SEQL);
        const unsigned lbase = lds_addr32(As);
#pragma unroll
        for (int i = 0; i < 8; ++i) {
            const unsigned off = (unsigned)(wave * 4096 + i * 512 + lane * 16);
            async_g2l_b128(lbase + off, gsrc + off);
        }
        wait_asynccnt0();
    }
    __syncthreads();

    const int nt = wave;                         // 0..7
    v8f acc = {};
    for (int k = 0; k < SEQL; k += 32) {
        v16bf a  = load_tile(As,  SEQL, 0,       k, lane);   // LDS
        v16bf bm = load_tile(Wih, SEQL, nt * 16, k, lane);   // global (L2-resident)
        acc = __builtin_amdgcn_wmma_f32_16x16x32_bf16(false, a, false, bm,
                                                      (short)0, acc, false, false);
    }
    const int n = nt * 16 + (lane & 15);
    const int mbase = mt * 16 + ((lane >> 4) & 1) * 8;
    const float bias = b_ih[n];
#pragma unroll
    for (int v = 0; v < 8; ++v)
        xproj[(size_t)(mbase + v) * HID + n] = acc[v] + bias;
}

// ---------------- kernel 3: tanh RNN scan over channels ----------------
// Recurrence independent per batch row: grid(4), block 256 (8 waves).
// Block handles batch rows [16*bx, 16*bx+16); wave w owns N-tile w.
__global__ void __launch_bounds__(256) k_rnn(const float* __restrict__ xproj,          // [512*64,128]
                                             const unsigned short* __restrict__ Whh,   // [128,128] bf16
                                             const float* __restrict__ b_hh,
                                             unsigned short* __restrict__ hs) {        // [(b*CH+c),128] bf16
    __shared__ unsigned short hbf[16 * HID];      // current hidden state (bf16)
    __shared__ unsigned short wsh[HID * HID];     // W_hh staged in LDS

    const int tid = threadIdx.x, lane = tid & 31, wave = tid >> 5;
    const int b0 = blockIdx.x * 16;

    // stage W_hh (32 KB) via async copy: 8 waves x 8 issues x 512 B
    {
        const unsigned lbase = lds_addr32(wsh);
#pragma unroll
        for (int i = 0; i < 8; ++i) {
            const unsigned off = (unsigned)(wave * 4096 + i * 512 + lane * 16);
            async_g2l_b128(lbase + off, (const char*)Whh + off);
        }
        wait_asynccnt0();
    }
    for (int i = tid; i < 16 * HID / 2; i += 256)
        ((unsigned int*)hbf)[i] = 0u;
    __syncthreads();

    const int n = wave * 16 + (lane & 15);
    const float bh = b_hh[n];
    const int mloc = ((lane >> 4) & 1) * 8;

    for (int c = 0; c < CH; ++c) {
        v8f acc = {};
#pragma unroll
        for (int k = 0; k < HID; k += 32) {
            v16bf a  = load_tile(hbf, HID, 0,         k, lane);
            v16bf bm = load_tile(wsh, HID, wave * 16, k, lane);
            acc = __builtin_amdgcn_wmma_f32_16x16x32_bf16(false, a, false, bm,
                                                          (short)0, acc, false, false);
        }
        __syncthreads();   // everyone done reading hbf
#pragma unroll
        for (int v = 0; v < 8; ++v) {
            const int bl = mloc + v;           // local batch row 0..15
            const int bg = b0 + bl;            // global batch row
            const float val = tanhf(acc[v] + xproj[((size_t)c * BATCH + bg) * HID + n] + bh);
            const unsigned short bf = f2bf(val);
            hbf[bl * HID + n] = bf;
            hs[((size_t)bg * CH + c) * HID + n] = bf;
        }
        __syncthreads();   // hbf updated before next step reads it
    }
}

// ---------------- kernel 4: trend GEMM + seasonal GEMM + de-norm ----------------
// Block = one 16-row M-strip; trend strip (32 KB) + hs strip (4 KB) async-staged
// to LDS once and shared by all 21 N-tiles (wave w handles nt = w, w+8, w+16).
// Epilogue transposes each 16x16 tile through padded LDS so stores are
// 64B-contiguous along the channel dimension of y.
__global__ void __launch_bounds__(256) k_out(const unsigned short* __restrict__ Tr,   // [32768,1024]
                                             const unsigned short* __restrict__ Wtr,  // [336,1024]
                                             const unsigned short* __restrict__ Hs,   // [32768,128]
                                             const unsigned short* __restrict__ Wfc,  // [336,128]
                                             const float* __restrict__ b_tr,
                                             const float* __restrict__ b_fc,
                                             const float* __restrict__ stdev,
                                             const float* __restrict__ means,
                                             float* __restrict__ y) {                 // [64,336,512]
    __shared__ unsigned short Ts[16 * SEQL];     // 32 KB trend strip
    __shared__ unsigned short Ss[16 * HID];      // 4 KB hs strip
    __shared__ float tbuf[8][16 * 17];           // per-wave padded tile (8.5 KB)
    const int lane = threadIdx.x & 31;
    const int wave = threadIdx.x >> 5;
    const int mt = blockIdx.x;                   // 0..2047

    {
        const char* gt = (const char*)(Tr + (size_t)mt * 16 * SEQL);
        const char* gh = (const char*)(Hs + (size_t)mt * 16 * HID);
        const unsigned lt = lds_addr32(Ts);
        const unsigned lh = lds_addr32(Ss);
#pragma unroll
        for (int i = 0; i < 8; ++i) {            // 32 KB trend
            const unsigned off = (unsigned)(wave * 4096 + i * 512 + lane * 16);
            async_g2l_b128(lt + off, gt + off);
        }
        {                                        // 4 KB hs: one issue per wave
            const unsigned off = (unsigned)(wave * 512 + lane * 16);
            async_g2l_b128(lh + off, gh + off);
        }
        wait_asynccnt0();
    }
    __syncthreads();

    // this M-strip covers rows r = mt*16 .. mt*16+15  (r = b*CH + c, c contiguous)
    const int r0 = mt * 16;
    const int b  = r0 >> 9;                      // batch index (strip never crosses b)
    const int c0 = r0 & (CH - 1);                // first channel of strip
    float* ybase = y + (size_t)b * PRED * CH + c0;

    for (int nt = wave; nt < 21; nt += 8) {      // wave-uniform loop, EXEC stays full
        v8f acc = {};
        for (int k = 0; k < SEQL; k += 32) {                  // trend path, K=1024
            v16bf a  = load_tile(Ts,  SEQL, 0,       k, lane);
            v16bf bm = load_tile(Wtr, SEQL, nt * 16, k, lane);
            acc = __builtin_amdgcn_wmma_f32_16x16x32_bf16(false, a, false, bm,
                                                          (short)0, acc, false, false);
        }
#pragma unroll
        for (int k = 0; k < HID; k += 32) {                   // seasonal path, K=128
            v16bf a  = load_tile(Ss,  HID, 0,       k, lane);
            v16bf bm = load_tile(Wfc, HID, nt * 16, k, lane);
            acc = __builtin_amdgcn_wmma_f32_16x16x32_bf16(false, a, false, bm,
                                                          (short)0, acc, false, false);
        }
        // fold bias + de-normalization, stash tile into per-wave padded LDS
        const int p_lane = lane & 15;                          // N index within tile
        const float bias = b_tr[nt * 16 + p_lane] + b_fc[nt * 16 + p_lane];
        const int mhalf = ((lane >> 4) & 1) * 8;
        float* tw = tbuf[wave];
#pragma unroll
        for (int v = 0; v < 8; ++v) {
            const int ml = mhalf + v;                          // M index = channel
            const int r  = r0 + ml;
            tw[ml * 17 + p_lane] = (acc[v] + bias) * stdev[r] + means[r];
        }
        // coalesced emit: lanes scan channels fast (64B contiguous per 16 lanes)
        const int cl = lane & 15;                              // channel within strip
        const int ph = lane >> 4;                              // p sub-row 0/1
        float* yp = ybase + (size_t)(nt * 16) * CH + cl;
#pragma unroll
        for (int j = 0; j < 8; ++j) {
            const int pl = ph + 2 * j;                         // p within tile
            yp[(size_t)pl * CH] = tw[cl * 17 + pl];
        }
    }
}

// ---------------- launch ----------------
extern "C" void kernel_launch(void* const* d_in, const int* in_sizes, int n_in,
                              void* d_out, int out_size, void* d_ws, size_t ws_size,
                              hipStream_t stream) {
    const float* x_enc = (const float*)d_in[0];
    const float* W_ih  = (const float*)d_in[4];
    const float* W_hh  = (const float*)d_in[5];
    const float* b_ih  = (const float*)d_in[6];
    const float* b_hh  = (const float*)d_in[7];
    const float* W_fc  = (const float*)d_in[8];
    const float* b_fc  = (const float*)d_in[9];
    const float* W_tr  = (const float*)d_in[10];
    const float* b_tr  = (const float*)d_in[11];
    float* y = (float*)d_out;

    // workspace partitioning (256B aligned)
    char* ws = (char*)d_ws;
    size_t off = 0;
    auto alloc = [&](size_t bytes) {
        void* p = ws + off;
        off += (bytes + 255) & ~(size_t)255;
        return p;
    };
    const size_t M = (size_t)BATCH * CH;                      // 32768
    unsigned short* seas_bf  = (unsigned short*)alloc(M * SEQL * 2);   // 64 MB
    unsigned short* trend_bf = (unsigned short*)alloc(M * SEQL * 2);   // 64 MB
    unsigned short* hs_bf    = (unsigned short*)alloc(M * HID * 2);    // 8 MB
    float*          xproj    = (float*)alloc(M * HID * 4);             // 16 MB
    float*          means    = (float*)alloc(M * 4);
    float*          stdev    = (float*)alloc(M * 4);
    unsigned short* Wih_bf   = (unsigned short*)alloc((size_t)HID * SEQL * 2);
    unsigned short* Whh_bf   = (unsigned short*)alloc((size_t)HID * HID * 2);
    unsigned short* Wfc_bf   = (unsigned short*)alloc((size_t)PRED * HID * 2);
    unsigned short* Wtr_bf   = (unsigned short*)alloc((size_t)PRED * SEQL * 2);

    // 0) weight conversion to bf16
    auto cvt = [&](const float* s, unsigned short* d, int n) {
        k_cvt_bf16<<<(n + 255) / 256, 256, 0, stream>>>(s, d, n);
    };
    cvt(W_ih, Wih_bf, HID * SEQL);
    cvt(W_hh, Whh_bf, HID * HID);
    cvt(W_fc, Wfc_bf, PRED * HID);
    cvt(W_tr, Wtr_bf, PRED * SEQL);

    // 1) normalize + decompose
    k_norm_decomp<<<dim3(CH / 256, BATCH), 256, 0, stream>>>(x_enc, means, stdev,
                                                             seas_bf, trend_bf);

    // 2) xproj GEMM: one block per 16-row M-strip
    k_xproj<<<2048, 256, 0, stream>>>(seas_bf, Wih_bf, b_ih, xproj);

    // 3) RNN scan over channels (4 blocks = 4 independent batch-row groups)
    k_rnn<<<4, 256, 0, stream>>>(xproj, Whh_bf, b_hh, hs_bf);

    // 4) fused output GEMMs + de-normalization: one block per 16-row M-strip
    k_out<<<2048, 256, 0, stream>>>(trend_bf, Wtr_bf, hs_bf, Wfc_bf,
                                    b_tr, b_fc, stdev, means, y);
}
